// VQVAE_20315195310125
// MI455X (gfx1250) — matrix operand order repaired and checked by reading, not compiled, not run on hardware
//
#include <hip/hip_runtime.h>
#include <hip/hip_bf16.h>

// ---------------- types / helpers ----------------
typedef __attribute__((ext_vector_type(16))) _Float16 v16h;
typedef __attribute__((ext_vector_type(8)))  float    v8f;
typedef _Float16 half_t;

union Frag16 { v16h v; _Float16 h[16]; unsigned u[8]; };

__device__ __forceinline__ v8f wmma_f16f32(const Frag16& a, const Frag16& b, v8f c) {
  // D = A(16x32 f16) * B(32x16 f16) + C(16x16 f32)
  return __builtin_amdgcn_wmma_f32_16x16x32_f16(false, a.v, false, b.v, (short)0, c,
                                                false, false);
}

// A-operand k order used here (contiguous, same as B loads): VGPR v, lane-half hi -> starting K of the packed pair
__device__ __forceinline__ int a_kbase(int v, int hi) {
  return 2 * v + 16 * hi;
}

__device__ __forceinline__ v8f zero8() {
  v8f z = {0.f, 0.f, 0.f, 0.f, 0.f, 0.f, 0.f, 0.f};
  return z;
}

// Intra-wave LDS producer->consumer ordering (DS ops are in-order per wave;
// fence stops compiler reordering and drains DScnt).
__device__ __forceinline__ void wave_lds_fence() {
  __builtin_amdgcn_wave_barrier();
  asm volatile("s_wait_dscnt 0" ::: "memory");
  __builtin_amdgcn_wave_barrier();
}

// ---------------- problem constants ----------------
#define T_     100
#define STEPS_ 50
#define HE_    64
#define E_     128
#define HD_    256
#define NE_    512
#define K2_    192   // conv2 im2col K = 3*64, ordered K = k*64 + ci
#define BETA_  0.25f

// ---------------- prep kernels ----------------
__global__ void k_zero(float* scal) {
  if (blockIdx.x == 0 && threadIdx.x < 4) scal[threadIdx.x] = 0.f;
}

__global__ void k_prep(const float* c2w, const float* cb, const float* d1w,
                       const float* d2w, const float* d3w, const float* imask,
                       half_t* W2a, half_t* cb16, float* cbnorm,
                       half_t* d1w16, half_t* d2w16, half_t* d3w16,
                       float* scal, int M) {
  int stride = gridDim.x * blockDim.x;
  int tid0 = blockIdx.x * blockDim.x + threadIdx.x;
  // conv2 weights -> A-matrix ordering W2a[oc][k*64+ci]
  for (int i = tid0; i < HE_ * K2_; i += stride) {
    int oc = i / K2_, rem = i % K2_, k = rem / HE_, ci = rem % HE_;
    W2a[i] = (half_t)c2w[oc * K2_ + ci * 3 + k];
  }
  for (int i = tid0; i < NE_ * E_; i += stride) cb16[i] = (half_t)cb[i];
  for (int i = tid0; i < NE_; i += stride) {
    float s = 0.f;
    for (int e = 0; e < E_; ++e) { float v = cb[i * E_ + e]; s += v * v; }
    cbnorm[i] = s;
  }
  for (int i = tid0; i < HD_ * E_; i += stride) d1w16[i] = (half_t)d1w[i];
  for (int i = tid0; i < HD_ * HD_; i += stride) d2w16[i] = (half_t)d2w[i];
  for (int i = tid0; i < 112 * HD_; i += stride) {  // pad rows 100..111 with 0
    int t = i / HD_, e = i % HD_;
    d3w16[i] = (half_t)(t < T_ ? d3w[t * HD_ + e] : 0.f);
  }
  float s = 0.f;
  for (int i = tid0; i < M; i += stride) s += imask[i];
  if (s != 0.f) atomicAdd(&scal[3], s);
}

// ---------------- encoder: conv1(VALU, sliding window) + conv2(WMMA) + mean + lat ----
// 2 waves/block, one instance per wave.
__launch_bounds__(64)
__global__ void k_encode(const float* __restrict__ x, const float* __restrict__ tmask,
                         const float* __restrict__ imask,
                         const float* __restrict__ c1w, const float* __restrict__ c1b,
                         const float* __restrict__ c2b,
                         const float* __restrict__ latw, const float* __restrict__ latb,
                         const half_t* __restrict__ W2a_g,
                         half_t* __restrict__ zf16, float* __restrict__ znorm, int M) {
  __shared__ alignas(16) half_t W2a[HE_ * K2_];     // 24 KB, shared by block
  __shared__ alignas(16) float  xw_s[2][T_ + 4];    // +4: window overrun slack
  __shared__ alignas(16) half_t h1p_s[2][66 * HE_]; // [l+1][ci] f16, zero-padded
  __shared__ alignas(16) float  hbar_s[2][HE_];

  int tid = threadIdx.x;
  for (int i = tid; i < HE_ * K2_; i += blockDim.x) W2a[i] = W2a_g[i];
  __syncthreads();

  int wave = tid >> 5, lane = tid & 31;
  int inst = blockIdx.x * 2 + wave;
  if (inst >= M) return;
  float im = imask[inst];
  if (im <= 0.f) {                          // z = (...)*im == 0: short-circuit
    for (int j = lane; j < E_; j += 32) zf16[(size_t)inst * E_ + j] = (half_t)0.f;
    if (lane == 0) znorm[inst] = 0.f;
    return;
  }
  float*  xw   = xw_s[wave];
  half_t* h1p  = h1p_s[wave];
  float*  hbar = hbar_s[wave];

  const float* xr = x + (size_t)inst * T_;
  const float* tr = tmask + (size_t)inst * T_;
  for (int j = lane; j < T_; j += 32) xw[j] = xr[j] * tr[j] * im;
  if (lane < 4) xw[T_ + lane] = 0.f;        // slack so the window never reads junk
  for (int j = lane; j < 66 * HE_; j += 32) h1p[j] = (half_t)0.f;
  wave_lds_fence();

  // conv1: 2 out-channels per lane, sliding window over stride-2 pairs.
  // Window for output l covers pairs ls = l-1, l, l+1.
  for (int rep = 0; rep < 2; ++rep) {
    int oc = lane + 32 * rep;
    float w00 = c1w[(oc * 2 + 0) * 3 + 0], w01 = c1w[(oc * 2 + 0) * 3 + 1],
          w02 = c1w[(oc * 2 + 0) * 3 + 2];
    float w10 = c1w[(oc * 2 + 1) * 3 + 0], w11 = c1w[(oc * 2 + 1) * 3 + 1],
          w12 = c1w[(oc * 2 + 1) * 3 + 2];
    float b = c1b[oc];
    float x0 = 0.f, x1 = 0.f;               // pair at ls = -1 (zero pad)
    float2 p0 = *(const float2*)&xw[0];     // pair at ls = 0
    float2 p1 = *(const float2*)&xw[2];     // pair at ls = 1
    float x2 = p0.x, x3 = p0.y, x4 = p1.x, x5 = p1.y;
    for (int l = 0; l < STEPS_; ++l) {
      float s = b + w00 * x0 + w10 * x1 + w01 * x2 + w11 * x3 + w02 * x4 + w12 * x5;
      s = s > 0.f ? s : 0.f;
      h1p[(l + 1) * HE_ + oc] = (half_t)s;
      x0 = x2; x1 = x3; x2 = x4; x3 = x5;
      float2 pn = *(const float2*)&xw[2 * l + 4];   // pair at ls = l+2 (slack-padded)
      x4 = (l + 2 < STEPS_) ? pn.x : 0.f;
      x5 = (l + 2 < STEPS_) ? pn.y : 0.f;
    }
  }
  wave_lds_fence();

  // conv2 as WMMA GEMM: out(64 x 50pad64) = W2a(64 x 192) * col(192 x 50pad64)
  int n = lane & 15, hi = lane >> 4;
  float mean[4][8];
#pragma unroll
  for (int mt = 0; mt < 4; ++mt)
#pragma unroll
    for (int r = 0; r < 8; ++r) mean[mt][r] = 0.f;

  for (int nt = 0; nt < 4; ++nt) {
    v8f acc[4];
#pragma unroll
    for (int mt = 0; mt < 4; ++mt) acc[mt] = zero8();
#pragma unroll
    for (int kc = 0; kc < 6; ++kc) {
      int k = kc >> 1;
      int cibase = (kc & 1) * 32 + hi * 16;
      Frag16 bf;
      int li = nt * 16 + n + k;            // padded index (l + k), l = nt*16+n
#pragma unroll
      for (int v = 0; v < 8; ++v) {
        int ci = cibase + 2 * v;
        bf.u[v] = *(const unsigned*)&h1p[li * HE_ + ci];
      }
#pragma unroll
      for (int mt = 0; mt < 4; ++mt) {
        Frag16 af;
        int oc = mt * 16 + n;
#pragma unroll
        for (int v = 0; v < 8; ++v)
          af.u[v] = *(const unsigned*)&W2a[oc * K2_ + kc * 32 + a_kbase(v, hi)];
        acc[mt] = wmma_f16f32(af, bf, acc[mt]);
      }
    }
    int l = nt * 16 + n;
    bool lvalid = l < STEPS_;
#pragma unroll
    for (int mt = 0; mt < 4; ++mt)
#pragma unroll
      for (int r = 0; r < 8; ++r) {
        int oc = mt * 16 + hi * 8 + r;
        float val = acc[mt][r] + c2b[oc];
        val = val > 0.f ? val : 0.f;
        if (lvalid) mean[mt][r] += val;
      }
  }
  // reduce over the 16 column lanes, write hbar
#pragma unroll
  for (int mt = 0; mt < 4; ++mt)
#pragma unroll
    for (int r = 0; r < 8; ++r) {
      float s = mean[mt][r];
      for (int mk = 1; mk < 16; mk <<= 1) s += __shfl_xor(s, mk, 32);
      if (n == 0) hbar[mt * 16 + hi * 8 + r] = s * (1.f / STEPS_);
    }
  wave_lds_fence();

  // lat: z = hbar @ lat_w.T + lat_b (VALU; 64-long dots)
  float zpart = 0.f;
  half_t* zrow = zf16 + (size_t)inst * E_;
  for (int j = 0; j < 4; ++j) {
    int e = lane + 32 * j;
    float s = latb[e];
    for (int he = 0; he < HE_; ++he) s += hbar[he] * latw[e * HE_ + he];
    s *= im;
    zrow[e] = (half_t)s;
    zpart += s * s;
  }
  for (int mk = 1; mk < 32; mk <<= 1) zpart += __shfl_xor(zpart, mk, 32);
  if (lane == 0) znorm[inst] = zpart;
}

// ---------------- VQ: distances via WMMA, argmin, loss, gather z_q ----------------
// 4 waves/block, 16 rows/wave.
__launch_bounds__(128)
__global__ void k_vq(const half_t* __restrict__ zf16, const float* __restrict__ znorm,
                     const float* __restrict__ imask,
                     const half_t* __restrict__ cb16, const float* __restrict__ cbnorm,
                     half_t* __restrict__ zq16, float* __restrict__ outIdx,
                     float* scal) {
  int wave = threadIdx.x >> 5, lane = threadIdx.x & 31;
  int n = lane & 15, hi = lane >> 4;
  int row0 = (blockIdx.x * 4 + wave) * 16;

  Frag16 af[4];
  const half_t* zr = zf16 + (size_t)(row0 + n) * E_;
#pragma unroll
  for (int kc = 0; kc < 4; ++kc)
#pragma unroll
    for (int v = 0; v < 8; ++v)
      af[kc].u[v] = *(const unsigned*)&zr[kc * 32 + a_kbase(v, hi)];

  float minv[8];
  int   mini[8];
#pragma unroll
  for (int r = 0; r < 8; ++r) { minv[r] = 3.4e38f; mini[r] = 0; }

  for (int nt = 0; nt < NE_ / 16; ++nt) {
    v8f acc = zero8();
    Frag16 bf;
    const half_t* cbr = cb16 + (size_t)(nt * 16 + n) * E_;
#pragma unroll
    for (int kc = 0; kc < 4; ++kc) {
#pragma unroll
      for (int v = 0; v < 8; ++v)
        bf.u[v] = *(const unsigned*)&cbr[kc * 32 + hi * 16 + 2 * v];
      acc = wmma_f16f32(af[kc], bf, acc);
    }
    float cn = cbnorm[nt * 16 + n];
    int code = nt * 16 + n;
#pragma unroll
    for (int r = 0; r < 8; ++r) {
      float d = cn - 2.f * acc[r];
      if (d < minv[r]) { minv[r] = d; mini[r] = code; }   // first-min on ties
    }
  }
  // argmin reduction across the 16 column lanes (keep smallest index on ties)
#pragma unroll
  for (int r = 0; r < 8; ++r) {
    float v = minv[r]; int i = mini[r];
    for (int mk = 1; mk < 16; mk <<= 1) {
      float ov = __shfl_xor(v, mk, 32);
      int   oi = __shfl_xor(i, mk, 32);
      if (ov < v || (ov == v && oi < i)) { v = ov; i = oi; }
    }
    minv[r] = v; mini[r] = i;
  }
  // per-row: index output + vq loss
  if (n == 0) {
    float vqpart = 0.f;
#pragma unroll
    for (int r = 0; r < 8; ++r) {
      int m = row0 + hi * 8 + r;
      float im = imask[m];
      outIdx[m] = (im > 0.f) ? (float)mini[r] : -1.f;
      // ||z-c||^2 = ||z||^2 - 2 z.c + ||c||^2 = znorm + minval
      float loss = (1.f + BETA_) * (znorm[m] + minv[r]) * (1.f / E_);
      vqpart += loss * im;
    }
    atomicAdd(&scal[2], vqpart);
  }
  // gather z_q = codebook[idx] * im (straight-through == codebook entry)
#pragma unroll
  for (int r = 0; r < 8; ++r) {
    int m = row0 + hi * 8 + r;
    float im = imask[m];
    const half_t* crow = cb16 + (size_t)mini[r] * E_;
    half_t* qrow = zq16 + (size_t)m * E_;
    if (im > 0.f) {
#pragma unroll
      for (int j = 0; j < 8; ++j) qrow[n * 8 + j] = crow[n * 8 + j];
    } else {
#pragma unroll
      for (int j = 0; j < 8; ++j) qrow[n * 8 + j] = (half_t)0.f;
    }
  }
}

// ---------------- decoder: 3 fused WMMA GEMM stages + masked MSE ----------------
// 2 waves/block, 16 rows/wave; intermediates in per-wave LDS.
__launch_bounds__(64)
__global__ void k_decode(const half_t* __restrict__ zq16,
                         const half_t* __restrict__ d1w16, const float* __restrict__ d1b,
                         const half_t* __restrict__ d2w16, const float* __restrict__ d2b,
                         const half_t* __restrict__ d3w16, const float* __restrict__ d3b,
                         const float* __restrict__ x, const float* __restrict__ tmask,
                         const float* __restrict__ imask,
                         float* __restrict__ xhat, float* scal) {
  __shared__ alignas(16) half_t h2a_s[2][16 * HD_];   // 8 KB per wave
  __shared__ alignas(16) half_t h2b_s[2][16 * HD_];
  int wave = threadIdx.x >> 5, lane = threadIdx.x & 31;
  int n = lane & 15, hi = lane >> 4;
  int row0 = (blockIdx.x * 2 + wave) * 16;
  half_t* h2a = h2a_s[wave];
  half_t* h2b = h2b_s[wave];

  Frag16 af[8];
  // ---- stage 1: h2a = relu(z_q @ d1_w.T + d1_b), K=128 ----
  const half_t* zr = zq16 + (size_t)(row0 + n) * E_;
#pragma unroll
  for (int kc = 0; kc < 4; ++kc)
#pragma unroll
    for (int v = 0; v < 8; ++v)
      af[kc].u[v] = *(const unsigned*)&zr[kc * 32 + a_kbase(v, hi)];
  for (int ot = 0; ot < HD_ / 16; ++ot) {
    v8f acc = zero8();
    int o = ot * 16 + n;
    Frag16 bf;
    const half_t* wr = d1w16 + (size_t)o * E_;
#pragma unroll
    for (int kc = 0; kc < 4; ++kc) {
#pragma unroll
      for (int v = 0; v < 8; ++v)
        bf.u[v] = *(const unsigned*)&wr[kc * 32 + hi * 16 + 2 * v];
      acc = wmma_f16f32(af[kc], bf, acc);
    }
    float b = d1b[o];
#pragma unroll
    for (int r = 0; r < 8; ++r) {
      float val = acc[r] + b;
      val = val > 0.f ? val : 0.f;
      h2a[(hi * 8 + r) * HD_ + o] = (half_t)val;
    }
  }
  wave_lds_fence();

  // ---- stage 2: h2b = relu(h2a @ d2_w.T + d2_b), K=256 ----
#pragma unroll
  for (int kc = 0; kc < 8; ++kc)
#pragma unroll
    for (int v = 0; v < 8; ++v)
      af[kc].u[v] = *(const unsigned*)&h2a[n * HD_ + kc * 32 + a_kbase(v, hi)];
  for (int ot = 0; ot < HD_ / 16; ++ot) {
    v8f acc = zero8();
    int o = ot * 16 + n;
    Frag16 bf;
    const half_t* wr = d2w16 + (size_t)o * HD_;
#pragma unroll
    for (int kc = 0; kc < 8; ++kc) {
#pragma unroll
      for (int v = 0; v < 8; ++v)
        bf.u[v] = *(const unsigned*)&wr[kc * 32 + hi * 16 + 2 * v];
      acc = wmma_f16f32(af[kc], bf, acc);
    }
    float b = d2b[o];
#pragma unroll
    for (int r = 0; r < 8; ++r) {
      float val = acc[r] + b;
      val = val > 0.f ? val : 0.f;
      h2b[(hi * 8 + r) * HD_ + o] = (half_t)val;
    }
  }
  wave_lds_fence();

  // ---- stage 3: x_hat = h2b @ d3_w.T + d3_b (100 cols, padded to 112) ----
#pragma unroll
  for (int kc = 0; kc < 8; ++kc)
#pragma unroll
    for (int v = 0; v < 8; ++v)
      af[kc].u[v] = *(const unsigned*)&h2b[n * HD_ + kc * 32 + a_kbase(v, hi)];
  float rnum = 0.f, rden = 0.f;
  for (int ot = 0; ot < 7; ++ot) {
    v8f acc = zero8();
    int t = ot * 16 + n;
    Frag16 bf;
    const half_t* wr = d3w16 + (size_t)t * HD_;   // rows 100..111 zero-padded
#pragma unroll
    for (int kc = 0; kc < 8; ++kc) {
#pragma unroll
      for (int v = 0; v < 8; ++v)
        bf.u[v] = *(const unsigned*)&wr[kc * 32 + hi * 16 + 2 * v];
      acc = wmma_f16f32(af[kc], bf, acc);
    }
    if (t < T_) {
      float b = d3b[t];
#pragma unroll
      for (int r = 0; r < 8; ++r) {
        int m = row0 + hi * 8 + r;
        float val = acc[r] + b;
        xhat[(size_t)m * T_ + t] = val;
        float w = imask[m] * tmask[(size_t)m * T_ + t];
        float diff = val - x[(size_t)m * T_ + t];
        rnum += diff * diff * w;
        rden += w;
      }
    }
  }
  for (int mk = 1; mk < 32; mk <<= 1) {
    rnum += __shfl_xor(rnum, mk, 32);
    rden += __shfl_xor(rden, mk, 32);
  }
  if (lane == 0) { atomicAdd(&scal[0], rnum); atomicAdd(&scal[1], rden); }
}

// ---------------- finalize ----------------
__global__ void k_final(const float* scal, float* out_scalars) {
  if (blockIdx.x == 0 && threadIdx.x == 0) {
    float den = scal[1] > 1.f ? scal[1] : 1.f;
    out_scalars[0] = scal[0] / den;                 // recon_loss
    float is = scal[3] > 1.f ? scal[3] : 1.f;
    out_scalars[1] = scal[2] / is;                  // vq_reduced
  }
}

// ---------------- host launch ----------------
extern "C" void kernel_launch(void* const* d_in, const int* in_sizes, int n_in,
                              void* d_out, int out_size, void* d_ws, size_t ws_size,
                              hipStream_t stream) {
  const float* x     = (const float*)d_in[0];
  const float* tmask = (const float*)d_in[1];
  const float* imask = (const float*)d_in[2];
  const float* c1w   = (const float*)d_in[3];
  const float* c1b   = (const float*)d_in[4];
  const float* c2w   = (const float*)d_in[5];
  const float* c2b   = (const float*)d_in[6];
  const float* latw  = (const float*)d_in[7];
  const float* latb  = (const float*)d_in[8];
  const float* cb    = (const float*)d_in[9];
  const float* d1w   = (const float*)d_in[10];
  const float* d1b   = (const float*)d_in[11];
  const float* d2w   = (const float*)d_in[12];
  const float* d2b   = (const float*)d_in[13];
  const float* d3w   = (const float*)d_in[14];
  const float* d3b   = (const float*)d_in[15];

  const int M = in_sizes[2];                       // B*N = 51200
  float* out = (float*)d_out;
  float* xhat        = out;                        // M*T floats
  float* out_scalars = out + (size_t)M * T_;       // recon_loss, vq_reduced
  float* out_idx     = out + (size_t)M * T_ + 2;   // M indices (as float)

  // workspace carve-out (256B aligned)
  char* ws = (char*)d_ws;
  size_t off = 0;
  auto carve = [&](size_t bytes) -> void* {
    off = (off + 255) & ~(size_t)255;
    void* p = ws + off;
    off += bytes;
    return p;
  };
  float*  scal   = (float*)carve(4 * sizeof(float));
  half_t* W2a    = (half_t*)carve((size_t)HE_ * K2_ * sizeof(half_t));
  half_t* cb16   = (half_t*)carve((size_t)NE_ * E_ * sizeof(half_t));
  float*  cbnorm = (float*)carve((size_t)NE_ * sizeof(float));
  half_t* d1w16  = (half_t*)carve((size_t)HD_ * E_ * sizeof(half_t));
  half_t* d2w16  = (half_t*)carve((size_t)HD_ * HD_ * sizeof(half_t));
  half_t* d3w16  = (half_t*)carve((size_t)112 * HD_ * sizeof(half_t));
  half_t* zf16   = (half_t*)carve((size_t)M * E_ * sizeof(half_t));
  half_t* zq16   = (half_t*)carve((size_t)M * E_ * sizeof(half_t));
  float*  znorm  = (float*)carve((size_t)M * sizeof(float));
  (void)ws_size; (void)n_in; (void)out_size;

  k_zero<<<1, 32, 0, stream>>>(scal);
  k_prep<<<256, 256, 0, stream>>>(c2w, cb, d1w, d2w, d3w, imask,
                                  W2a, cb16, cbnorm, d1w16, d2w16, d3w16, scal, M);
  k_encode<<<M / 2, 64, 0, stream>>>(x, tmask, imask, c1w, c1b, c2b, latw, latb,
                                     W2a, zf16, znorm, M);
  k_vq<<<M / 64, 128, 0, stream>>>(zf16, znorm, imask, cb16, cbnorm,
                                   zq16, out_idx, scal);
  k_decode<<<M / 32, 64, 0, stream>>>(zq16, d1w16, d1b, d2w16, d2b, d3w16, d3b,
                                      x, tmask, imask, xhat, scal);
  k_final<<<1, 32, 0, stream>>>(scal, out_scalars);
}
